// BatchAll_61624190763673
// MI455X (gfx1250) — compile-verified, matrix operand order checked by reading, not run on hardware
//
#include <hip/hip_runtime.h>
#include <hip/hip_bf16.h>

typedef __attribute__((ext_vector_type(16))) __bf16 v16bf;
typedef __attribute__((ext_vector_type(2)))  __bf16 v2bf;
typedef __attribute__((ext_vector_type(8)))  float  v8f;

constexpr int   kB      = 512;          // batch
constexpr int   kD      = 1024;         // embedding dim
constexpr int   kKT     = kD / 32;      // k-tiles of 32
constexpr float kMargin = 0.2f;
constexpr float kEps    = 1e-16f;

// ---------------------------------------------------------------------------
// Prep: fp32 [kB][kD] -> hi/lo bf16 split, stored in WMMA-fragment order:
//   frag index f = tile*kKT + kt ; storage[(f*32 + lane)*16 + elem]
// CDNA5 16-bit A/B layout: lanes 0-15 hold row=tile*16+lane,
//   elems 0-7 -> K=kt*32+{0..7}, elems 8-15 -> K=kt*32+{16..23};
// lanes 16-31: same rows, K bases shifted by +8.
// Each thread handles one elem-pair -> dst index == 2*pid (coalesced b32).
// ---------------------------------------------------------------------------
__global__ __launch_bounds__(256) void prep_kernel(
    const float* __restrict__ src, __bf16* __restrict__ hi,
    __bf16* __restrict__ lo) {
  const int pid = blockIdx.x * 256 + threadIdx.x;   // [0, kB*kD/2)
  const int ep   = pid & 7;          // elem pair within fragment row
  const int lane = (pid >> 3) & 31;
  const int frag = pid >> 8;         // tile * kKT + kt
  const int tile = frag / kKT;
  const int kt   = frag % kKT;

  const int m    = tile * 16 + (lane & 15);
  const int half = lane >> 4;
  const int e    = ep * 2;
  const int kk   = (e & 7) + 8 * half + 16 * (e >> 3);
  const float* p = src + (size_t)m * kD + kt * 32 + kk;

  const float f0 = p[0], f1 = p[1];
  const __bf16 h0 = (__bf16)f0, h1 = (__bf16)f1;
  const __bf16 l0 = (__bf16)(f0 - (float)h0);
  const __bf16 l1 = (__bf16)(f1 - (float)h1);
  *(v2bf*)(hi + (size_t)pid * 2) = (v2bf){h0, h1};
  *(v2bf*)(lo + (size_t)pid * 2) = (v2bf){l0, l1};
}

// ---------------------------------------------------------------------------
// GEMM: pairwise = img @ snt^T via bf16 hi/lo-split WMMA (~fp32 accuracy).
// One wave computes a 32x32 output block (2x2 tiles, 4 accumulators);
// 12 WMMAs per k-step interleaved across accumulators to hide WMMA hazards.
// ---------------------------------------------------------------------------
__device__ __forceinline__ v16bf load_frag(const __bf16* __restrict__ base,
                                           int tile, int kt, int lane) {
  return *(const v16bf*)(base +
                         (((size_t)tile * kKT + kt) * 32 + lane) * 16);
}

#define WMMA_BF16(A, Bm, C)                                                   \
  __builtin_amdgcn_wmma_f32_16x16x32_bf16(false, (A), false, (Bm), (short)0,  \
                                          (C), false, false)

__global__ __launch_bounds__(32) void triplet_gemm_kernel(
    const __bf16* __restrict__ hiA, const __bf16* __restrict__ loA,
    const __bf16* __restrict__ hiS, const __bf16* __restrict__ loS,
    float* __restrict__ pairwise) {
  const int lane = threadIdx.x;          // full wave32, EXEC all ones
  const int tm0  = blockIdx.x * 2;       // 16-row tile indices
  const int tn0  = blockIdx.y * 2;

  v8f acc00 = {}, acc01 = {}, acc10 = {}, acc11 = {};

  for (int kt = 0; kt < kKT; ++kt) {
    const v16bf a0h = load_frag(hiA, tm0,     kt, lane);
    const v16bf a1h = load_frag(hiA, tm0 + 1, kt, lane);
    const v16bf a0l = load_frag(loA, tm0,     kt, lane);
    const v16bf a1l = load_frag(loA, tm0 + 1, kt, lane);
    const v16bf b0h = load_frag(hiS, tn0,     kt, lane);
    const v16bf b1h = load_frag(hiS, tn0 + 1, kt, lane);
    const v16bf b0l = load_frag(loS, tn0,     kt, lane);
    const v16bf b1l = load_frag(loS, tn0 + 1, kt, lane);

    // hi*hi
    acc00 = WMMA_BF16(a0h, b0h, acc00);
    acc01 = WMMA_BF16(a0h, b1h, acc01);
    acc10 = WMMA_BF16(a1h, b0h, acc10);
    acc11 = WMMA_BF16(a1h, b1h, acc11);
    // hi*lo
    acc00 = WMMA_BF16(a0h, b0l, acc00);
    acc01 = WMMA_BF16(a0h, b1l, acc01);
    acc10 = WMMA_BF16(a1h, b0l, acc10);
    acc11 = WMMA_BF16(a1h, b1l, acc11);
    // lo*hi  (lo*lo is below fp32 ulp)
    acc00 = WMMA_BF16(a0l, b0h, acc00);
    acc01 = WMMA_BF16(a0l, b1h, acc01);
    acc10 = WMMA_BF16(a1l, b0h, acc10);
    acc11 = WMMA_BF16(a1l, b1h, acc11);
  }

  // C/D layout: lanes 0-15 -> N=lane, VGPR r -> M=r; lanes 16-31 -> M=r+8
  const int cl = lane & 15;
  const int rh = (lane >> 4) * 8;
  v8f accs[2][2] = {{acc00, acc01}, {acc10, acc11}};
#pragma unroll
  for (int i = 0; i < 2; ++i) {
#pragma unroll
    for (int j = 0; j < 2; ++j) {
      const int col   = (tn0 + j) * 16 + cl;
      const int rbase = (tm0 + i) * 16 + rh;
#pragma unroll
      for (int r = 0; r < 8; ++r)
        pairwise[(size_t)(rbase + r) * kB + col] = accs[i][j][r];
    }
  }
}

// ---------------------------------------------------------------------------
// One block per anchor a. Compact positives (label==label[a], expected ~4),
// then accumulate hinge over negatives x positives. Deterministic: serial
// compaction by thread 0, tree reductions, no atomics.
// ---------------------------------------------------------------------------
__global__ __launch_bounds__(256) void triplet_reduce_kernel(
    const int* __restrict__ labels, const float* __restrict__ pairwise,
    float* __restrict__ partial) {
  __shared__ int   slab[kB];
  __shared__ float srow[kB];
  __shared__ float spos[kB];
  __shared__ int   snpos;
  __shared__ float rsum[256];
  __shared__ float rcnt[256];

  const int a   = blockIdx.x;
  const int tid = threadIdx.x;

  for (int j = tid; j < kB; j += 256) {
    slab[j] = labels[j];
    srow[j] = pairwise[(size_t)a * kB + j];
  }
  __syncthreads();

  if (tid == 0) {
    const int la = slab[a];
    int c = 0;
    for (int j = 0; j < kB; ++j)
      if (slab[j] == la) spos[c++] = srow[j];  // includes p==a, as reference
    snpos = c;
  }
  __syncthreads();

  const int la = slab[a];
  const int np = snpos;
  float sum = 0.f, cnt = 0.f;
  for (int n = tid; n < kB; n += 256) {
    if (slab[n] != la) {                        // valid negative
      const float dn = srow[n];
      for (int p = 0; p < np; ++p) {
        const float v = spos[p] - dn + kMargin;
        sum += fmaxf(v, 0.f);
        if (v > kEps) cnt += 1.f;
      }
    }
  }

  rsum[tid] = sum;
  rcnt[tid] = cnt;
  __syncthreads();
  for (int s = 128; s > 0; s >>= 1) {
    if (tid < s) {
      rsum[tid] += rsum[tid + s];
      rcnt[tid] += rcnt[tid + s];
    }
    __syncthreads();
  }
  if (tid == 0) {
    partial[a]      = rsum[0];
    partial[kB + a] = rcnt[0];
  }
}

__global__ __launch_bounds__(256) void triplet_final_kernel(
    const float* __restrict__ partial, float* __restrict__ out) {
  __shared__ float rsum[256];
  __shared__ float rcnt[256];
  const int tid = threadIdx.x;
  float s = 0.f, c = 0.f;
  for (int j = tid; j < kB; j += 256) {
    s += partial[j];
    c += partial[kB + j];
  }
  rsum[tid] = s;
  rcnt[tid] = c;
  __syncthreads();
  for (int st = 128; st > 0; st >>= 1) {
    if (tid < st) {
      rsum[tid] += rsum[tid + st];
      rcnt[tid] += rcnt[tid + st];
    }
    __syncthreads();
  }
  if (tid == 0) out[0] = rsum[0] / (rcnt[0] + kEps);
}

// ---------------------------------------------------------------------------
extern "C" void kernel_launch(void* const* d_in, const int* in_sizes, int n_in,
                              void* d_out, int out_size, void* d_ws, size_t ws_size,
                              hipStream_t stream) {
  const int*   labels = (const int*)d_in[0];
  const float* img    = (const float*)d_in[1];
  const float* snt    = (const float*)d_in[2];

  char* ws = (char*)d_ws;
  float* pairwise = (float*)ws;                                   // 1 MB
  float* partial  = (float*)(ws + (size_t)kB * kB * 4);           // 4 KB
  __bf16* hiA = (__bf16*)(ws + (size_t)kB * kB * 4 + 4096);       // 1 MB each
  __bf16* loA = hiA + (size_t)kB * kD;
  __bf16* hiS = loA + (size_t)kB * kD;
  __bf16* loS = hiS + (size_t)kB * kD;

  const int prepBlocks = (kB * kD / 2) / 256;                     // 1024
  prep_kernel<<<prepBlocks, 256, 0, stream>>>(img, hiA, loA);
  prep_kernel<<<prepBlocks, 256, 0, stream>>>(snt, hiS, loS);

  triplet_gemm_kernel<<<dim3(kB / 32, kB / 32), 32, 0, stream>>>(
      hiA, loA, hiS, loS, pairwise);
  triplet_reduce_kernel<<<kB, 256, 0, stream>>>(labels, pairwise, partial);
  triplet_final_kernel<<<1, 256, 0, stream>>>(partial, (float*)d_out);
}
